// RegionProposalNetwork_34978213658703
// MI455X (gfx1250) — matrix operand order repaired and checked by reading, not compiled, not run on hardware
//
#include <hip/hip_runtime.h>

// ---------------------------------------------------------------------------
// RPN on MI455X (gfx1250).
//  Pass P: fp32 -> bf16 pack (zero-padded input halo, transposed W1) ~9us HBM.
//  Pass 1: implicit-GEMM 3x3 conv via v_wmma_f32_16x16x32_bf16, double-
//          buffered global_load_async_to_lds_b128 staging (ASYNCcnt).
//  Pass 2: 1x1 conv head (WMMA) + sigmoid/roi epilogue.
//  Pass 3: cropped anchors + image assignments (elementwise).
// Fallback conv (no extra ws) kept for small ws_size.
// ---------------------------------------------------------------------------

typedef __attribute__((ext_vector_type(16))) __bf16 v16bf;
typedef __attribute__((ext_vector_type(8)))  __bf16 v8bf;
typedef __attribute__((ext_vector_type(8)))  float  v8f;
typedef __attribute__((ext_vector_type(4)))  float  v4f;
typedef __attribute__((ext_vector_type(4)))  unsigned v4u;

#define HH   128
#define WW   128
#define BB   8
#define CIN  256
#define KTOT 2304                 // 9 * 256
#define MTOT (BB * HH * WW)       // 131072
#define N1   (MTOT * 9)           // 1179648
#define NC   (KTOT / 64)          // 36 staged chunks

#define LDA  72                   // A LDS row stride (elems), 144B
#define LDB  80                   // B LDS row stride (elems), 160B
#define LDW  264                  // W2 LDS row stride (elems), 528B
#define ABYTES (128 * LDA * 2)    // one A stage buffer, bytes
#define BBYTES (128 * LDB * 2)    // one B stage buffer, bytes

#define PADPIX (130 * 130)                  // 16900 padded pixels per image
#define INPAD_ELEMS (BB * PADPIX * CIN)     // 34,611,200 bf16

static __device__ __forceinline__ unsigned short f2bf(float f) {
  unsigned u = __float_as_uint(f);
  u += 0x7FFFu + ((u >> 16) & 1u);          // round-to-nearest-even
  return (unsigned short)(u >> 16);
}

static __device__ __forceinline__ v16bf frag16(const unsigned short* p0,
                                               const unsigned short* p1) {
  v8bf lo = *(const v8bf*)p0;
  v8bf hi = *(const v8bf*)p1;
  return __builtin_shufflevector(lo, hi, 0,1,2,3,4,5,6,7,8,9,10,11,12,13,14,15);
}

static __device__ __forceinline__ unsigned lds_off(const void* p) {
  // generic LDS pointer = {aperture_hi32, lds_offset32}; low 32 bits = offset
  return (unsigned)(unsigned long long)p;
}

// ---------------------------------------------------------------------------
// Pack input: fp32 NHWC -> bf16 [8][130][130][256] with 1-pixel zero halo.
// ---------------------------------------------------------------------------
__global__ __launch_bounds__(256)
void pack_input(const float* __restrict__ in, unsigned short* __restrict__ inPad) {
  int gid = blockIdx.x * 256 + threadIdx.x;      // one 8-elem (16B) chunk each
  int pixel = gid >> 5;                          // 32 chunks per pixel
  int c8    = (gid & 31) << 3;
  int b   = pixel / PADPIX;
  int rem = pixel - b * PADPIX;
  int yp  = rem / 130;
  int xp  = rem - yp * 130;
  unsigned o0 = 0, o1 = 0, o2 = 0, o3 = 0;
  if (yp >= 1 && yp <= 128 && xp >= 1 && xp <= 128) {
    const float* p = in + (((size_t)(b * 128 + (yp - 1)) * 128) + (xp - 1)) * 256 + c8;
    v4f a = *(const v4f*)p;
    v4f c = *(const v4f*)(p + 4);
    o0 = (unsigned)f2bf(a[0]) | ((unsigned)f2bf(a[1]) << 16);
    o1 = (unsigned)f2bf(a[2]) | ((unsigned)f2bf(a[3]) << 16);
    o2 = (unsigned)f2bf(c[0]) | ((unsigned)f2bf(c[1]) << 16);
    o3 = (unsigned)f2bf(c[2]) | ((unsigned)f2bf(c[3]) << 16);
  }
  v4u o = {o0, o1, o2, o3};
  *(v4u*)&inPad[(size_t)gid * 8] = o;
}

// ---------------------------------------------------------------------------
// Pack W1: fp32 [2304][256] (K-major HWIO) -> bf16 transposed [256 n][2304 k].
// ---------------------------------------------------------------------------
__global__ __launch_bounds__(256)
void pack_w1(const float* __restrict__ W1, unsigned short* __restrict__ W1T) {
  int gid = blockIdx.x * 256 + threadIdx.x;      // 2304*64 vec4 tasks
  int k  = gid >> 6;
  int n4 = (gid & 63) << 2;
  v4f v = *(const v4f*)(W1 + (size_t)k * 256 + n4);
  #pragma unroll
  for (int j = 0; j < 4; ++j)
    W1T[(size_t)(n4 + j) * KTOT + k] = f2bf(v[j]);
}

// ---------------------------------------------------------------------------
// Shared inner compute: one K=64 chunk (2 WMMA k-steps), fragments preloaded
// into distinct registers so ds_load latency is batched, WMMAs back-to-back.
// ---------------------------------------------------------------------------
static __device__ __forceinline__ void chunk_mma(const unsigned short* Ab,
                                                 const unsigned short* Bb,
                                                 int wm, int wn, int rsel,
                                                 int bhalf, v8f acc[2][4]) {
  #pragma unroll
  for (int kk = 0; kk < 2; ++kk) {
    const int kb = kk * 32;
    v16bf af[2];
    v16bf bfr[4];
    #pragma unroll
    for (int tm = 0; tm < 2; ++tm) {
      const unsigned short* pa =
          &Ab[(wm * 32 + tm * 16 + rsel) * LDA + kb + bhalf * 8];
      af[tm] = frag16(pa, pa + 16);
    }
    #pragma unroll
    for (int tn = 0; tn < 4; ++tn) {
      const unsigned short* pb =
          &Bb[(wn * 64 + tn * 16 + rsel) * LDB + kb + bhalf * 8];
      bfr[tn] = frag16(pb, pb + 16);
    }
    #pragma unroll
    for (int tn = 0; tn < 4; ++tn)
      #pragma unroll
      for (int tm = 0; tm < 2; ++tm)
        acc[tm][tn] = __builtin_amdgcn_wmma_f32_16x16x32_bf16(
            false, af[tm], false, bfr[tn], (short)0, acc[tm][tn], false, false);
  }
}

// ---------------------------------------------------------------------------
// Kernel 1 (async): conv 3x3 + bias + ReLU as implicit GEMM, bf16 WMMA.
//   Block tile 128M x 128N, wave tile 32M x 64N, K staged 64/chunk,
//   double-buffered global_load_async_to_lds_b128.
// ---------------------------------------------------------------------------
__global__ __launch_bounds__(256)
void rpn_conv3x3_wmma_async(const unsigned short* __restrict__ inPad,
                            const unsigned short* __restrict__ W1T,
                            const float* __restrict__ b1,
                            unsigned short* __restrict__ feats) {
  __shared__ unsigned short Alds[2 * 128 * LDA];
  __shared__ unsigned short Blds[2 * 128 * LDB];

  const int tid  = threadIdx.x;
  const int lane = tid & 31;
  const int wave = tid >> 5;
  const int wm   = wave & 3;
  const int wn   = wave >> 2;
  const int m0   = blockIdx.x * 128;
  const int n0   = blockIdx.y * 128;
  const int rsel  = lane & 15;
  const int bhalf = lane >> 4;

  // per-thread staging invariants: 4 A copies + 4 B copies per chunk
  int pixBase[4];                      // padded pixel index (no tap offset)
  int cOff[4];                         // cin sub-offset within chunk
  unsigned ldsA[4], ldsB[4];
  const unsigned short* gB[4];
  #pragma unroll
  for (int j = 0; j < 4; ++j) {
    int q   = tid + j * 256;           // 0..1023
    int row = q >> 3;                  // 0..127
    int seg = q & 7;                   // 16B segment within 128B row
    int m   = m0 + row;
    int b   = m >> 14;
    int rem = m & 16383;
    int y   = rem >> 7;
    int x   = rem & 127;
    pixBase[j] = (b * 130 + y) * 130 + x;
    cOff[j]    = seg * 8;
    ldsA[j]    = lds_off(&Alds[row * LDA + seg * 8]);
    gB[j]      = W1T + (size_t)(n0 + row) * KTOT + seg * 8;
    ldsB[j]    = lds_off(&Blds[row * LDB + seg * 8]);
  }

  v8f acc[2][4];
  #pragma unroll
  for (int tm = 0; tm < 2; ++tm)
    #pragma unroll
    for (int tn = 0; tn < 4; ++tn)
      acc[tm][tn] = v8f{};

  // ---- stage chunk 0 into buffer 0 ----
  #pragma unroll
  for (int j = 0; j < 4; ++j) {
    const unsigned short* ga = inPad + (size_t)pixBase[j] * 256 + cOff[j];
    asm volatile("global_load_async_to_lds_b128 %0, %1, off"
                 :: "v"(ldsA[j]), "v"(ga) : "memory");
    asm volatile("global_load_async_to_lds_b128 %0, %1, off"
                 :: "v"(ldsB[j]), "v"(gB[j]) : "memory");
  }

  #pragma unroll 1
  for (int c = 0; c < NC; ++c) {
    const int buf = c & 1;

    if (c + 1 < NC) {               // issue next stage into other buffer
      const int kc  = (c + 1) * 64;
      const int tap = kc >> 8;
      const int ky  = tap / 3;
      const int kx  = tap - ky * 3;
      const int tapOff = ky * 130 + kx;
      const int cb  = kc & 255;
      const int nb  = buf ^ 1;
      #pragma unroll
      for (int j = 0; j < 4; ++j) {
        const unsigned short* ga =
            inPad + (size_t)(pixBase[j] + tapOff) * 256 + cb + cOff[j];
        asm volatile("global_load_async_to_lds_b128 %0, %1, off"
                     :: "v"(ldsA[j] + nb * ABYTES), "v"(ga) : "memory");
        const unsigned short* gb = gB[j] + kc;
        asm volatile("global_load_async_to_lds_b128 %0, %1, off"
                     :: "v"(ldsB[j] + nb * BBYTES), "v"(gb) : "memory");
      }
      asm volatile("s_wait_asynccnt 8" ::: "memory");   // stage c complete
    } else {
      asm volatile("s_wait_asynccnt 0" ::: "memory");
    }
    __syncthreads();

    chunk_mma(&Alds[buf * 128 * LDA], &Blds[buf * 128 * LDB],
              wm, wn, rsel, bhalf, acc);
    __syncthreads();
  }

  // ---- epilogue: bias + ReLU -> bf16 feats[m][n] ----
  const int rhalf = bhalf * 8;
  #pragma unroll
  for (int tn = 0; tn < 4; ++tn) {
    const int ng   = n0 + wn * 64 + tn * 16 + rsel;
    const float bv = b1[ng];
    #pragma unroll
    for (int tm = 0; tm < 2; ++tm) {
      #pragma unroll
      for (int vg = 0; vg < 8; ++vg) {
        int   m = m0 + wm * 32 + tm * 16 + rhalf + vg;
        float v = acc[tm][tn][vg] + bv;
        v = v > 0.f ? v : 0.f;
        feats[(size_t)m * 256 + ng] = f2bf(v);
      }
    }
  }
}

// ---------------------------------------------------------------------------
// Kernel 1 (fallback, small ws): conversion-in-kernel version.
// ---------------------------------------------------------------------------
__global__ __launch_bounds__(256)
void rpn_conv3x3_wmma_legacy(const float* __restrict__ in,
                             const float* __restrict__ W1,
                             const float* __restrict__ b1,
                             unsigned short* __restrict__ feats) {
  __shared__ unsigned short Alds[128 * LDA];
  __shared__ unsigned short Blds[128 * LDB];

  const int tid  = threadIdx.x;
  const int lane = tid & 31;
  const int wave = tid >> 5;
  const int wm   = wave & 3;
  const int wn   = wave >> 2;
  const int m0   = blockIdx.x * 128;
  const int n0   = blockIdx.y * 128;
  const int rsel  = lane & 15;
  const int bhalf = lane >> 4;

  v8f acc[2][4];
  #pragma unroll
  for (int tm = 0; tm < 2; ++tm)
    #pragma unroll
    for (int tn = 0; tn < 4; ++tn)
      acc[tm][tn] = v8f{};

  #pragma unroll 1
  for (int chunk = 0; chunk < NC; ++chunk) {
    const int kc  = chunk * 64;
    const int tap = kc >> 8;
    const int ky  = tap / 3 - 1;
    const int kx  = tap % 3 - 1;
    const int cb  = kc & 255;

    #pragma unroll
    for (int i = 0; i < 8; ++i) {
      int q = tid + i * 256, row = q >> 4, c4 = (q & 15) << 2;
      int m = m0 + row, b = m >> 14, rem = m & 16383;
      int y = (rem >> 7) + ky, x = (rem & 127) + kx;
      v4f v = v4f{};
      if ((unsigned)y < 128u && (unsigned)x < 128u)
        v = *(const v4f*)(in + (((size_t)(b * 128 + y) * 128 + x) * 256 + cb + c4));
      unsigned p0 = (unsigned)f2bf(v[0]) | ((unsigned)f2bf(v[1]) << 16);
      unsigned p1 = (unsigned)f2bf(v[2]) | ((unsigned)f2bf(v[3]) << 16);
      *(unsigned long long*)&Alds[row * LDA + c4] =
          ((unsigned long long)p1 << 32) | p0;
    }
    #pragma unroll
    for (int i = 0; i < 8; ++i) {
      int q = tid + i * 256, kl = q >> 5, n4 = (q & 31) << 2;
      v4f v = *(const v4f*)(W1 + ((size_t)(kc + kl) * 256 + n0 + n4));
      #pragma unroll
      for (int j = 0; j < 4; ++j)
        Blds[(n4 + j) * LDB + kl] = f2bf(v[j]);
    }
    __syncthreads();
    chunk_mma(Alds, Blds, wm, wn, rsel, bhalf, acc);
    __syncthreads();
  }

  const int rhalf = bhalf * 8;
  #pragma unroll
  for (int tn = 0; tn < 4; ++tn) {
    const int ng   = n0 + wn * 64 + tn * 16 + rsel;
    const float bv = b1[ng];
    #pragma unroll
    for (int tm = 0; tm < 2; ++tm)
      #pragma unroll
      for (int vg = 0; vg < 8; ++vg) {
        int   m = m0 + wm * 32 + tm * 16 + rhalf + vg;
        float v = acc[tm][tn][vg] + bv;
        v = v > 0.f ? v : 0.f;
        feats[(size_t)m * 256 + ng] = f2bf(v);
      }
  }
}

// ---------------------------------------------------------------------------
// Kernel 2: 1x1 conv (256 -> 45, padded 48) + sigmoid/roi epilogue.
// ---------------------------------------------------------------------------
__global__ __launch_bounds__(256)
void rpn_head_wmma(const unsigned short* __restrict__ feats,
                   const float* __restrict__ W2,
                   const float* __restrict__ b2,
                   float* __restrict__ pred,
                   float* __restrict__ rois) {
  __shared__ unsigned short Wlds[48 * LDW];

  const int tid = threadIdx.x;
  for (int idx = tid; idx < 256 * 48; idx += 256) {
    int k = idx / 48, n = idx % 48;
    Wlds[n * LDW + k] = f2bf((n < 45) ? W2[k * 45 + n] : 0.f);
  }
  __syncthreads();

  const int lane  = tid & 31;
  const int wave  = tid >> 5;
  const int m0    = blockIdx.x * 128 + wave * 16;
  const int rsel  = lane & 15;
  const int bhalf = lane >> 4;

  v8f acc[3];
  #pragma unroll
  for (int t = 0; t < 3; ++t) acc[t] = v8f{};

  const unsigned short* arow = feats + (size_t)(m0 + rsel) * 256;
  #pragma unroll
  for (int ks = 0; ks < 8; ++ks) {
    const int kb = ks * 32;
    const unsigned short* pa = arow + kb + bhalf * 8;
    v16bf af = frag16(pa, pa + 16);
    v16bf bfr[3];
    #pragma unroll
    for (int t = 0; t < 3; ++t) {
      const unsigned short* pb = &Wlds[(t * 16 + rsel) * LDW + kb + bhalf * 8];
      bfr[t] = frag16(pb, pb + 16);
    }
    #pragma unroll
    for (int t = 0; t < 3; ++t)
      acc[t] = __builtin_amdgcn_wmma_f32_16x16x32_bf16(
          false, af, false, bfr[t], (short)0, acc[t], false, false);
  }

  #pragma unroll
  for (int t = 0; t < 3; ++t) {
    int j = t * 16 + rsel;
    if (j < 45) {
      float bias = b2[j];
      int a = j / 5, comp = j % 5;
      float s  = 460.8f * (float)(1 << (a / 3));
      int  rt  = a % 3;
      float th = (rt == 1) ? s : 0.5f * s;
      float tw = (rt == 2) ? s : 0.5f * s;
      #pragma unroll
      for (int vg = 0; vg < 8; ++vg) {
        int   m = m0 + bhalf * 8 + vg;
        float v = acc[t][vg] + bias;
        size_t r = (size_t)m * 9 + a;
        if (comp == 0) {
          pred[r] = 1.f / (1.f + __expf(-v));
        } else {
          int p = m & 16383;
          int ci = comp - 1;
          float anc;
          if (ci == 0)      anc = 8.f + 16.f * (float)(p & 127);
          else if (ci == 1) anc = 8.f + 16.f * (float)(p >> 7);
          else if (ci == 2) anc = th;
          else              anc = tw;
          rois[r * 4 + ci] = anc + v;
        }
      }
    }
  }
}

// ---------------------------------------------------------------------------
// Kernel 3: cropped anchors + image assignments.
// ---------------------------------------------------------------------------
__global__ __launch_bounds__(256)
void rpn_anchors(float* __restrict__ anch, int* __restrict__ assign) {
  int idx = blockIdx.x * 256 + threadIdx.x;
  if (idx >= N1) return;
  int m = idx / 9, a = idx - m * 9;
  int b = m >> 14;
  int p = m & 16383;
  float cy = 8.f + 16.f * (float)(p & 127);
  float cx = 8.f + 16.f * (float)(p >> 7);
  float s  = 460.8f * (float)(1 << (a / 3));
  int  rt  = a % 3;
  float th = (rt == 1) ? s : 0.5f * s;
  float tw = (rt == 2) ? s : 0.5f * s;
  float t  = fminf(fmaxf(cy - th, 0.f), 2048.f);
  float l  = fminf(fmaxf(cx - tw, 0.f), 2048.f);
  float bo = fminf(fmaxf(cy + th, 0.f), 2048.f);
  float rr = fminf(fmaxf(cx + tw, 0.f), 2048.f);
  v4f o = { 0.5f * (t + bo), 0.5f * (l + rr), 0.5f * (bo - t), 0.5f * (rr - l) };
  *(v4f*)(anch + (size_t)idx * 4) = o;
  assign[idx] = b;
}

// ---------------------------------------------------------------------------
extern "C" void kernel_launch(void* const* d_in, const int* in_sizes, int n_in,
                              void* d_out, int out_size, void* d_ws, size_t ws_size,
                              hipStream_t stream) {
  const float* in = (const float*)d_in[0];
  const float* W1 = (const float*)d_in[1];
  const float* b1 = (const float*)d_in[2];
  const float* W2 = (const float*)d_in[3];
  const float* b2 = (const float*)d_in[4];

  float* out    = (float*)d_out;
  float* pred   = out;
  float* rois   = out + (size_t)N1;
  float* anch   = out + (size_t)5 * N1;
  int*   assign = (int*)(out + (size_t)9 * N1);

  const size_t featsElems = (size_t)MTOT * 256;
  const size_t w1tElems   = (size_t)256 * KTOT;
  const size_t need_async =
      ((size_t)INPAD_ELEMS + featsElems + w1tElems) * sizeof(unsigned short);

  if (ws_size >= need_async) {
    unsigned short* inPad = (unsigned short*)d_ws;
    unsigned short* feats = inPad + (size_t)INPAD_ELEMS;
    unsigned short* W1T   = feats + featsElems;
    pack_input<<<dim3(INPAD_ELEMS / 8 / 256), 256, 0, stream>>>(in, inPad);
    pack_w1<<<dim3(KTOT * 64 / 256), 256, 0, stream>>>(W1, W1T);
    rpn_conv3x3_wmma_async<<<dim3(MTOT / 128, 2), 256, 0, stream>>>(inPad, W1T, b1, feats);
    rpn_head_wmma<<<dim3(MTOT / 128), 256, 0, stream>>>(feats, W2, b2, pred, rois);
  } else {
    unsigned short* feats = (unsigned short*)d_ws;
    rpn_conv3x3_wmma_legacy<<<dim3(MTOT / 128, 2), 256, 0, stream>>>(in, W1, b1, feats);
    rpn_head_wmma<<<dim3(MTOT / 128), 256, 0, stream>>>(feats, W2, b2, pred, rois);
  }
  rpn_anchors<<<dim3((N1 + 255) / 256), 256, 0, stream>>>(anch, assign);
}
